// EncoderDecoder_85126251807222
// MI455X (gfx1250) — matrix operand order repaired and checked by reading, not compiled, not run on hardware
//
#include <hip/hip_runtime.h>
#include <hip/hip_bf16.h>
#include <math.h>

// Problem dims
#define Bx 256
#define Tx 512
#define DINx 64
#define Hx 512
#define HDx 1024
#define QWx 64
#define QLx 64
#define QOUTx 4096
#define KBEAM 8
#define DELTAx 16

typedef __bf16 bf16_t;
typedef __attribute__((ext_vector_type(8)))  __bf16 bf16x8;
typedef __attribute__((ext_vector_type(16))) __bf16 bf16x16;
typedef __attribute__((ext_vector_type(8)))  float  f32x8;

union Frag16 {
    bf16x16 v;
    struct { bf16x8 lo; bf16x8 hi; } p;
};

// ---------------------------------------------------------------------------
// Elementwise helpers
// ---------------------------------------------------------------------------
__global__ void cast_f32_bf16(const float* __restrict__ s, __hip_bfloat16* __restrict__ d, long n) {
    long i = (long)blockIdx.x * blockDim.x + threadIdx.x;
    if (i < n) d[i] = __float2bfloat16(s[i]);
}
__global__ void zero_f32(float* __restrict__ p, long n) {
    long i = (long)blockIdx.x * blockDim.x + threadIdx.x;
    if (i < n) p[i] = 0.0f;
}
__global__ void zero_bf16(__hip_bfloat16* __restrict__ p, long n) {
    long i = (long)blockIdx.x * blockDim.x + threadIdx.x;
    if (i < n) p[i] = __float2bfloat16(0.0f);
}
__global__ void copy_bf16(const __hip_bfloat16* __restrict__ s, __hip_bfloat16* __restrict__ d, long n) {
    long i = (long)blockIdx.x * blockDim.x + threadIdx.x;
    if (i < n) d[i] = s[i];
}

// ---------------------------------------------------------------------------
// WMMA fragment loads
//   A (16xK, row-major): lane holds row lane%16; 8 K at (lane/16)*8, 8 at +16.
//   B (KxN from row-major W[N,K]): lane holds col lane%16; 8 K at (lane/16)*8,
//   8 at +16.
// ---------------------------------------------------------------------------
__device__ __forceinline__ void load_afrag(Frag16& f, const bf16_t* __restrict__ arow,
                                           int k, int lh) {
    f.p.lo = *(const bf16x8*)(arow + k + lh * 8);
    f.p.hi = *(const bf16x8*)(arow + k + 16 + lh * 8);
}
__device__ __forceinline__ void load_wfrag(Frag16& f, const bf16_t* __restrict__ wrow,
                                           int k, int lh) {
    f.p.lo = *(const bf16x8*)(wrow + k + lh * 8);
    f.p.hi = *(const bf16x8*)(wrow + k + 16 + lh * 8);
}

// ---------------------------------------------------------------------------
// Software-pipelined K-loop for one (A, W) pair: loads for step k+32 are
// issued before the WMMAs of step k so the wave never needs a full
// s_wait_loadcnt 0 in steady state; next-next W/A cachelines are prefetched
// (global_prefetch_b8).
// ---------------------------------------------------------------------------
__device__ __forceinline__ void wmma_pair_pipelined(
    const bf16_t* __restrict__ arow, const bf16_t* __restrict__ Wb,
    long wld, int tile_n, int K, int lm, int lh, f32x8 acc[4])
{
    const bf16_t* wrow0 = Wb + (size_t)(tile_n + 0 * 16 + lm) * wld;
    const bf16_t* wrow1 = Wb + (size_t)(tile_n + 1 * 16 + lm) * wld;
    const bf16_t* wrow2 = Wb + (size_t)(tile_n + 2 * 16 + lm) * wld;
    const bf16_t* wrow3 = Wb + (size_t)(tile_n + 3 * 16 + lm) * wld;

    Frag16 aC, w0C, w1C, w2C, w3C;
    load_afrag(aC, arow, 0, lh);
    load_wfrag(w0C, wrow0, 0, lh);
    load_wfrag(w1C, wrow1, 0, lh);
    load_wfrag(w2C, wrow2, 0, lh);
    load_wfrag(w3C, wrow3, 0, lh);

    for (int k = 32; k < K; k += 32) {
        Frag16 aN, w0N, w1N, w2N, w3N;
        load_afrag(aN, arow, k, lh);
        load_wfrag(w0N, wrow0, k, lh);
        load_wfrag(w1N, wrow1, k, lh);
        load_wfrag(w2N, wrow2, k, lh);
        load_wfrag(w3N, wrow3, k, lh);
        // Pull the following iteration's lines toward the caches.
        __builtin_prefetch(arow + k + 32, 0, 3);
        __builtin_prefetch(wrow0 + k + 32, 0, 3);
        __builtin_prefetch(wrow1 + k + 32, 0, 3);
        __builtin_prefetch(wrow2 + k + 32, 0, 3);
        __builtin_prefetch(wrow3 + k + 32, 0, 3);

        acc[0] = __builtin_amdgcn_wmma_f32_16x16x32_bf16(false, aC.v, false, w0C.v, (short)0, acc[0], false, false);
        acc[1] = __builtin_amdgcn_wmma_f32_16x16x32_bf16(false, aC.v, false, w1C.v, (short)0, acc[1], false, false);
        acc[2] = __builtin_amdgcn_wmma_f32_16x16x32_bf16(false, aC.v, false, w2C.v, (short)0, acc[2], false, false);
        acc[3] = __builtin_amdgcn_wmma_f32_16x16x32_bf16(false, aC.v, false, w3C.v, (short)0, acc[3], false, false);

        aC = aN; w0C = w0N; w1C = w1N; w2C = w2N; w3C = w3N;
    }
    acc[0] = __builtin_amdgcn_wmma_f32_16x16x32_bf16(false, aC.v, false, w0C.v, (short)0, acc[0], false, false);
    acc[1] = __builtin_amdgcn_wmma_f32_16x16x32_bf16(false, aC.v, false, w1C.v, (short)0, acc[1], false, false);
    acc[2] = __builtin_amdgcn_wmma_f32_16x16x32_bf16(false, aC.v, false, w2C.v, (short)0, acc[2], false, false);
    acc[3] = __builtin_amdgcn_wmma_f32_16x16x32_bf16(false, aC.v, false, w3C.v, (short)0, acc[3], false, false);
}

// ---------------------------------------------------------------------------
// WMMA GEMM:  C[M,N] = A[M,K] @ W[N,K]^T (+ A2[M,K2] @ W2[N,K2]^T) + bias0 + bias1
// A, W bf16 row-major; accumulate f32; output to Cf (f32) and/or Cb (bf16).
// One wave computes a 16(M) x 64(N) tile; block = 4 waves = 4 M-tiles.
// Requires: N % 64 == 0; K, K2 % 32 == 0; row starts 16B-aligned.
// ---------------------------------------------------------------------------
__global__ void __launch_bounds__(128)
gemm_bf16_wmma(const __hip_bfloat16* __restrict__ A,  long lda,
               const __hip_bfloat16* __restrict__ W,
               const __hip_bfloat16* __restrict__ A2, long lda2,
               const __hip_bfloat16* __restrict__ W2,
               const float* __restrict__ bias0, const float* __restrict__ bias1,
               float* __restrict__ Cf, __hip_bfloat16* __restrict__ Cb, long ldc,
               int M, int N, int K, int K2)
{
    const int wave   = threadIdx.x >> 5;
    const int lane   = threadIdx.x & 31;
    const int lm     = lane & 15;
    const int lh     = lane >> 4;
    const int tile_m = (blockIdx.y * 4 + wave) * 16;
    const int tile_n = blockIdx.x * 64;
    if (tile_m >= M) return;

    f32x8 acc[4];
#pragma unroll
    for (int t = 0; t < 4; ++t) {
        int n = tile_n + t * 16 + lm;
        float bv = 0.0f;
        if (bias0) bv += bias0[n];
        if (bias1) bv += bias1[n];
#pragma unroll
        for (int e = 0; e < 8; ++e) acc[t][e] = bv;
    }

    const bf16_t* arow = (const bf16_t*)A + (size_t)(tile_m + lm) * lda;
    wmma_pair_pipelined(arow, (const bf16_t*)W, (long)K, tile_n, K, lm, lh, acc);

    if (A2) {
        const bf16_t* arow2 = (const bf16_t*)A2 + (size_t)(tile_m + lm) * lda2;
        wmma_pair_pipelined(arow2, (const bf16_t*)W2, (long)K2, tile_n, K2, lm, lh, acc);
    }

    // C/D layout: lane -> N = lane%16; VGPR e -> M = e + 8*(lane/16)
#pragma unroll
    for (int t = 0; t < 4; ++t) {
        int n = tile_n + t * 16 + lm;
#pragma unroll
        for (int e = 0; e < 8; ++e) {
            int m = tile_m + e + 8 * lh;
            if (Cf) Cf[(size_t)m * ldc + n] = acc[t][e];
            if (Cb) Cb[(size_t)m * ldc + n] = __float2bfloat16(acc[t][e]);
        }
    }
}

// ---------------------------------------------------------------------------
// LSTM cell elementwise: gates (B x 4H, PyTorch order i,f,g,o) -> c, h.
// Writes h as bf16 (next GEMM input) and optionally into the bf16 sequence.
// ---------------------------------------------------------------------------
__device__ __forceinline__ float sigmoidf_(float x) { return 1.0f / (1.0f + __expf(-x)); }

__global__ void lstm_cell(const float* __restrict__ gates, float* __restrict__ c,
                          __hip_bfloat16* __restrict__ h_bf,
                          __hip_bfloat16* __restrict__ yseq, long ystride, int t)
{
    int idx = blockIdx.x * blockDim.x + threadIdx.x;
    if (idx >= Bx * Hx) return;
    int b = idx / Hx, j = idx - b * Hx;
    const float* g = gates + (size_t)b * 4 * Hx;
    float gi = sigmoidf_(g[j]);
    float gf = sigmoidf_(g[Hx + j]);
    float gg = tanhf(g[2 * Hx + j]);
    float go = sigmoidf_(g[3 * Hx + j]);
    float cv = gf * c[idx] + gi * gg;
    float hv = go * tanhf(cv);
    c[idx] = cv;
    h_bf[idx] = __float2bfloat16(hv);
    if (yseq) yseq[(size_t)b * ystride + (size_t)t * Hx + j] = __float2bfloat16(hv);
}

// ---------------------------------------------------------------------------
// Top-k (k<=8) over QOUT logits per row; one wave per row; lowest-index ties.
// Writes fw/fl indices and fills decoded[b][t][kk][{0,1}] = {fl, fw}.
// ---------------------------------------------------------------------------
__global__ void __launch_bounds__(32)
topk_kernel(const float* __restrict__ y, int k, int* __restrict__ fw, int* __restrict__ fl,
            float* __restrict__ decoded, int t)
{
    int b = blockIdx.x, lane = threadIdx.x;
    const float* row = y + (size_t)b * QOUTx;
    int chosen[KBEAM];
    for (int kk = 0; kk < k; ++kk) {
        float bestv = -__builtin_inff();
        int   besti = QOUTx;
        for (int i = lane; i < QOUTx; i += 32) {
            bool skip = false;
            for (int p = 0; p < kk; ++p) skip |= (chosen[p] == i);
            if (skip) continue;
            float v = row[i];
            if (v > bestv || (v == bestv && i < besti)) { bestv = v; besti = i; }
        }
        for (int off = 16; off > 0; off >>= 1) {
            float ov = __shfl_xor(bestv, off, 32);
            int   oi = __shfl_xor(besti, off, 32);
            if (ov > bestv || (ov == bestv && oi < besti)) { bestv = ov; besti = oi; }
        }
        chosen[kk] = besti;   // identical on all lanes after butterfly
        if (lane == 0) {
            int w = besti / QLx, l = besti % QLx;
            fw[b * KBEAM + kk] = w;
            fl[b * KBEAM + kk] = l;
            size_t base = (((size_t)b * DELTAx + t) * KBEAM + kk) * 2;
            decoded[base + 0] = (float)l;
            decoded[base + 1] = (float)w;
        }
    }
}

// ---------------------------------------------------------------------------
// emb = concat( sum_k (qw_w[:, fw_k] + qw_b), sum_k (ql_w[:, fl_k] + ql_b) )
// (one-hot @ W^T collapses to a column gather)
// ---------------------------------------------------------------------------
__global__ void embed_kernel(const int* __restrict__ fw, const int* __restrict__ fl, int k,
                             const float* __restrict__ qw_w, const float* __restrict__ qw_b,
                             const float* __restrict__ ql_w, const float* __restrict__ ql_b,
                             __hip_bfloat16* __restrict__ emb)
{
    int idx = blockIdx.x * blockDim.x + threadIdx.x;
    if (idx >= Bx * Hx) return;
    int b = idx / Hx, j = idx - b * Hx;
    float s = 0.0f;
    if (j < Hx / 2) {
        for (int kk = 0; kk < k; ++kk) s += qw_w[(size_t)j * QWx + fw[b * KBEAM + kk]];
        s += (float)k * qw_b[j];
    } else {
        int j2 = j - Hx / 2;
        for (int kk = 0; kk < k; ++kk) s += ql_w[(size_t)j2 * QLx + fl[b * KBEAM + kk]];
        s += (float)k * ql_b[j2];
    }
    emb[idx] = __float2bfloat16(s);
}

__global__ void avg_kernel(const float* __restrict__ dec, float* __restrict__ avg)
{
    int idx = blockIdx.x * blockDim.x + threadIdx.x;
    if (idx >= Bx * DELTAx * 2) return;
    int c = idx & 1;
    int t = (idx >> 1) % DELTAx;
    int b = idx / (DELTAx * 2);
    float s = 0.0f;
    for (int k = 0; k < KBEAM; ++k)
        s += dec[(((size_t)b * DELTAx + t) * KBEAM + k) * 2 + c];
    avg[idx] = s;
}

// ---------------------------------------------------------------------------
// Host orchestration
// ---------------------------------------------------------------------------
extern "C" void kernel_launch(void* const* d_in, const int* in_sizes, int n_in,
                              void* d_out, int out_size, void* d_ws, size_t ws_size,
                              hipStream_t stream)
{
    (void)in_sizes; (void)n_in; (void)out_size; (void)ws_size;

    const float* x       = (const float*)d_in[0];
    const float* efc1_w  = (const float*)d_in[1];
    const float* efc1_b  = (const float*)d_in[2];
    const float* efc2_w  = (const float*)d_in[3];
    const float* efc2_b  = (const float*)d_in[4];
    const float* efc3_w  = (const float*)d_in[5];
    const float* efc3_b  = (const float*)d_in[6];
    const float* el1_wih = (const float*)d_in[7];
    const float* el1_whh = (const float*)d_in[8];
    const float* el1_bih = (const float*)d_in[9];
    const float* el1_bhh = (const float*)d_in[10];
    const float* el2_wih = (const float*)d_in[11];
    const float* el2_whh = (const float*)d_in[12];
    const float* el2_bih = (const float*)d_in[13];
    const float* el2_bhh = (const float*)d_in[14];
    const float* dl1_wih = (const float*)d_in[15];
    const float* dl1_whh = (const float*)d_in[16];
    const float* dl1_bih = (const float*)d_in[17];
    const float* dl1_bhh = (const float*)d_in[18];
    const float* dl2_wih = (const float*)d_in[19];
    const float* dl2_whh = (const float*)d_in[20];
    const float* dl2_bih = (const float*)d_in[21];
    const float* dl2_bhh = (const float*)d_in[22];
    const float* dfc1_w  = (const float*)d_in[23];
    const float* dfc1_b  = (const float*)d_in[24];
    const float* dfc2_w  = (const float*)d_in[25];
    const float* dfc2_b  = (const float*)d_in[26];
    const float* dfc3_w  = (const float*)d_in[27];
    const float* dfc3_b  = (const float*)d_in[28];
    const float* qw_w    = (const float*)d_in[29];
    const float* qw_b    = (const float*)d_in[30];
    const float* ql_w    = (const float*)d_in[31];
    const float* ql_b    = (const float*)d_in[32];

    // ---- workspace bump allocator (requires ~325 MB of d_ws) ----
    char* wsp = (char*)d_ws;
    auto bump = [&](size_t bytes) -> void* {
        void* r = (void*)wsp;
        wsp += (bytes + 255) & ~(size_t)255;
        return r;
    };
    auto castw = [&](const float* src, size_t n) -> __hip_bfloat16* {
        __hip_bfloat16* d = (__hip_bfloat16*)bump(n * 2);
        cast_f32_bf16<<<(int)((n + 255) / 256), 256, 0, stream>>>(src, d, (long)n);
        return d;
    };

    __hip_bfloat16* xb   = (__hip_bfloat16*)bump((size_t)Bx * Tx * DINx * 2);
    __hip_bfloat16* bufA = (__hip_bfloat16*)bump((size_t)Bx * Tx * Hx * 2);
    __hip_bfloat16* bufB = (__hip_bfloat16*)bump((size_t)Bx * Tx * Hx * 2);

    __hip_bfloat16* w_efc1 = castw(efc1_w, (size_t)Hx * DINx);
    __hip_bfloat16* w_efc2 = castw(efc2_w, (size_t)Hx * Hx);
    __hip_bfloat16* w_efc3 = castw(efc3_w, (size_t)Hx * Hx);
    __hip_bfloat16* w_el1_wih = castw(el1_wih, (size_t)4 * Hx * Hx);
    __hip_bfloat16* w_el1_whh = castw(el1_whh, (size_t)4 * Hx * Hx);
    __hip_bfloat16* w_el2_wih = castw(el2_wih, (size_t)4 * Hx * Hx);
    __hip_bfloat16* w_el2_whh = castw(el2_whh, (size_t)4 * Hx * Hx);
    __hip_bfloat16* w_dl1_wih = castw(dl1_wih, (size_t)4 * Hx * Hx);
    __hip_bfloat16* w_dl1_whh = castw(dl1_whh, (size_t)4 * Hx * Hx);
    __hip_bfloat16* w_dl2_wih = castw(dl2_wih, (size_t)4 * Hx * Hx);
    __hip_bfloat16* w_dl2_whh = castw(dl2_whh, (size_t)4 * Hx * Hx);
    __hip_bfloat16* w_dfc1 = castw(dfc1_w, (size_t)HDx * Hx);
    __hip_bfloat16* w_dfc2 = castw(dfc2_w, (size_t)HDx * HDx);
    __hip_bfloat16* w_dfc3 = castw(dfc3_w, (size_t)QOUTx * HDx);

    float* c1 = (float*)bump((size_t)Bx * Hx * 4);
    float* c2 = (float*)bump((size_t)Bx * Hx * 4);
    __hip_bfloat16* h1b  = (__hip_bfloat16*)bump((size_t)Bx * Hx * 2);
    __hip_bfloat16* h2b  = (__hip_bfloat16*)bump((size_t)Bx * Hx * 2);
    float* gates         = (float*)bump((size_t)Bx * 4 * Hx * 4);
    __hip_bfloat16* encb = (__hip_bfloat16*)bump((size_t)Bx * Hx * 2);
    __hip_bfloat16* embb = (__hip_bfloat16*)bump((size_t)Bx * Hx * 2);
    __hip_bfloat16* f1b  = (__hip_bfloat16*)bump((size_t)Bx * HDx * 2);
    __hip_bfloat16* f2b  = (__hip_bfloat16*)bump((size_t)Bx * HDx * 2);
    float* ylog          = (float*)bump((size_t)Bx * QOUTx * 4);
    int* fwArr           = (int*)bump((size_t)Bx * KBEAM * 4);
    int* flArr           = (int*)bump((size_t)Bx * KBEAM * 4);

    auto gemm = [&](const __hip_bfloat16* A, long lda, const __hip_bfloat16* W,
                    const __hip_bfloat16* A2, long lda2, const __hip_bfloat16* W2,
                    const float* b0, const float* b1,
                    float* Cf, __hip_bfloat16* Cb, long ldc,
                    int M, int N, int K, int K2) {
        dim3 grid(N / 64, (M + 63) / 64);
        gemm_bf16_wmma<<<grid, 128, 0, stream>>>(A, lda, W, A2, lda2, W2,
                                                 b0, b1, Cf, Cb, ldc, M, N, K, K2);
    };

    float* out_decoded = (float*)d_out;                              // (B,16,8,2)
    float* out_avg     = out_decoded + (size_t)Bx * DELTAx * KBEAM * 2;

    // d_out is poisoned; decoded rows for t>=1, beams 1..7 must be zero.
    zero_f32<<<(Bx * DELTAx * KBEAM * 2 + Bx * DELTAx * 2 + 255) / 256, 256, 0, stream>>>(
        out_decoded, (long)(Bx * DELTAx * KBEAM * 2 + Bx * DELTAx * 2));

    // ---- encoder FC stack over all B*T tokens ----
    long nX = (long)Bx * Tx * DINx;
    cast_f32_bf16<<<(int)((nX + 255) / 256), 256, 0, stream>>>(x, xb, nX);
    gemm(xb, DINx, w_efc1, nullptr, 0, nullptr, efc1_b, nullptr,
         nullptr, bufA, Hx, Bx * Tx, Hx, DINx, 0);
    gemm(bufA, Hx, w_efc2, nullptr, 0, nullptr, efc2_b, nullptr,
         nullptr, bufB, Hx, Bx * Tx, Hx, Hx, 0);
    gemm(bufB, Hx, w_efc3, nullptr, 0, nullptr, efc3_b, nullptr,
         nullptr, bufA, Hx, Bx * Tx, Hx, Hx, 0);                      // e3 in bufA

    const int cellBlocks = (Bx * Hx + 255) / 256;

    // ---- encoder LSTM layer 1: reads e3 (bufA), writes y1 seq (bufB) ----
    zero_f32 <<<cellBlocks, 256, 0, stream>>>(c1, (long)Bx * Hx);
    zero_bf16<<<cellBlocks, 256, 0, stream>>>(h1b, (long)Bx * Hx);
    for (int t = 0; t < Tx; ++t) {
        gemm(bufA + (size_t)t * Hx, (long)Tx * Hx, w_el1_wih,
             h1b, Hx, w_el1_whh, el1_bih, el1_bhh,
             gates, nullptr, 4 * Hx, Bx, 4 * Hx, Hx, Hx);
        lstm_cell<<<cellBlocks, 256, 0, stream>>>(gates, c1, h1b, bufB, (long)Tx * Hx, t);
    }

    // ---- encoder LSTM layer 2: reads y1 (bufB); only final h needed ----
    zero_f32 <<<cellBlocks, 256, 0, stream>>>(c2, (long)Bx * Hx);
    zero_bf16<<<cellBlocks, 256, 0, stream>>>(h2b, (long)Bx * Hx);
    for (int t = 0; t < Tx; ++t) {
        gemm(bufB + (size_t)t * Hx, (long)Tx * Hx, w_el2_wih,
             h2b, Hx, w_el2_whh, el2_bih, el2_bhh,
             gates, nullptr, 4 * Hx, Bx, 4 * Hx, Hx, Hx);
        lstm_cell<<<cellBlocks, 256, 0, stream>>>(gates, c2, h2b, nullptr, 0, 0);
    }
    copy_bf16<<<cellBlocks, 256, 0, stream>>>(h2b, encb, (long)Bx * Hx);  // enc = e[:, -1]

    // ---- decoder: 16 autoregressive steps (states h1/c1, h2/c2 carry over) ----
    for (int t = 0; t < DELTAx; ++t) {
        const __hip_bfloat16* xi = (t == 0) ? encb : embb;
        gemm(xi, Hx, w_dl1_wih, h1b, Hx, w_dl1_whh, dl1_bih, dl1_bhh,
             gates, nullptr, 4 * Hx, Bx, 4 * Hx, Hx, Hx);
        lstm_cell<<<cellBlocks, 256, 0, stream>>>(gates, c1, h1b, nullptr, 0, 0);

        gemm(h1b, Hx, w_dl2_wih, h2b, Hx, w_dl2_whh, dl2_bih, dl2_bhh,
             gates, nullptr, 4 * Hx, Bx, 4 * Hx, Hx, Hx);
        lstm_cell<<<cellBlocks, 256, 0, stream>>>(gates, c2, h2b, nullptr, 0, 0);

        gemm(h2b, Hx, w_dfc1, nullptr, 0, nullptr, dfc1_b, nullptr,
             nullptr, f1b, HDx, Bx, HDx, Hx, 0);
        gemm(f1b, HDx, w_dfc2, nullptr, 0, nullptr, dfc2_b, nullptr,
             nullptr, f2b, HDx, Bx, HDx, HDx, 0);
        gemm(f2b, HDx, w_dfc3, nullptr, 0, nullptr, dfc3_b, nullptr,
             ylog, nullptr, QOUTx, Bx, QOUTx, HDx, 0);

        int k = (t == 0) ? KBEAM : 1;
        topk_kernel<<<Bx, 32, 0, stream>>>(ylog, k, fwArr, flArr, out_decoded, t);
        embed_kernel<<<cellBlocks, 256, 0, stream>>>(fwArr, flArr, k,
                                                     qw_w, qw_b, ql_w, ql_b, embb);
    }

    avg_kernel<<<(Bx * DELTAx * 2 + 255) / 256, 256, 0, stream>>>(out_decoded, out_avg);
}
